// WGDDetector_63848983823163
// MI455X (gfx1250) — matrix operand order repaired and checked by reading, not compiled, not run on hardware
//
#include <hip/hip_runtime.h>
#include <hip/hip_bf16.h>

// ---------------- problem constants (from reference) ----------------
#define DEPTH  17
#define NN     ((1 << DEPTH) - 1)        // 131071 nodes
#define NIN    ((1 << (DEPTH - 1)) - 1)  // 65535 internal nodes
#define NE     (NN - 1)                  // 131070 edges (parent-child pairs)
#define HID    64
#define HEADS  4
#define NODE_F 13
#define EIN    196                       // 64 + 64 + 4 + 64
#define EIN_PAD 224                      // padded to multiple of 32 for WMMA K
#define NEG_SLOPE 0.2f
#define LN_EPS 1e-5f

// max staged weight block: 64x256 (GAT projection) = 16384 halves = 32 KB LDS
#define B_LDS_HALVES 16384

typedef _Float16 h16;
typedef __attribute__((ext_vector_type(16))) _Float16 v16h;
typedef __attribute__((ext_vector_type(8)))  float    v8f;

// =====================================================================
// WMMA GEMM:  C[M x Ncols] = act( A[M x K] * B[K x Ncols](f16) + bias )
// AT in {float, h16}, CT in {float, h16}.  One wave owns a 16x64 tile.
//
// B (weights, <=32KB) is staged into LDS once per workgroup: kills the
// ~64 MB/GEMM of redundant per-wave B re-reads on the global path; the
// inner loop then feeds v_wmma_f32_16x16x32_f16 from strided 16-bit LDS loads.
//
// A fragment (16-bit A 16x32, ISA 7.12.2): lane&15 = M row;
//   lanes 0-15: halves 0..7 -> K 0..7,  halves 8..15 -> K 16..23
//   lanes16-31: halves 0..7 -> K 8..15, halves 8..15 -> K 24..31
// B fragment (f16 B 32x16): lane&15 = N col, halves follow the same K order as A.
// C fragment: VGPR r, lanes 0-15 -> M=r, lanes 16-31 -> M=8+r, N = lane&15.
// =====================================================================
template <typename AT, typename CT>
__global__ void k_gemm(const AT* __restrict__ A, int M, int K,
                       const h16* __restrict__ B, int Ncols,
                       const float* __restrict__ bias,
                       CT* __restrict__ C, int relu)
{
    __shared__ h16 Bs[B_LDS_HALVES];

    // ---- stage full B block (K x Ncols halves) into LDS ----
    {
        const uint4* src = (const uint4*)B;        // 8 halves per uint4
        uint4* dst = (uint4*)Bs;
        const int nvec = (K * Ncols) >> 3;
        for (int i = threadIdx.x; i < nvec; i += blockDim.x) dst[i] = src[i];
    }
    __syncthreads();   // s_barrier_signal / s_barrier_wait

    const int wave = (blockIdx.x * blockDim.x + threadIdx.x) >> 5;
    const int lane = threadIdx.x & 31;
    const int MT = (M + 15) >> 4;
    const int NB = Ncols >> 6;
    if (wave >= MT * NB) return;
    const int m0 = (wave % MT) << 4;
    const int n0 = (wave / MT) << 6;

    v8f acc[4] = {};

    const int rowA = m0 + (lane & 15);
    const int koff = (lane >> 4) << 3;        // 0 or 8

    for (int k0 = 0; k0 < K; k0 += 32) {
        v16h a = {};
        if (rowA < M) {
            const AT* ap = A + (size_t)rowA * K + k0 + koff;
#pragma unroll
            for (int t = 0; t < 8; ++t) {
                a[t]     = (h16)ap[t];
                a[t + 8] = (h16)ap[t + 16];
            }
            if (k0 + 32 < K)
                __builtin_prefetch((const void*)(ap + 32), 0, 1);  // global_prefetch_b8
        }
        // B fragments from LDS (16-bit loads, stride Ncols), then the WMMA burst
        const h16* bp = Bs + (size_t)(k0 + koff) * Ncols + n0 + (lane & 15);
#pragma unroll
        for (int nn = 0; nn < 4; ++nn) {
            v16h b;
#pragma unroll
            for (int t = 0; t < 8; ++t) {
                b[t]     = bp[(size_t)t * Ncols + nn * 16];
                b[t + 8] = bp[(size_t)(t + 16) * Ncols + nn * 16];
            }
            acc[nn] = __builtin_amdgcn_wmma_f32_16x16x32_f16(
                false, a, false, b, (short)0, acc[nn], false, false);
        }
    }

    const int mrow = m0 + ((lane >> 4) << 3);
    const int ncol = lane & 15;
#pragma unroll
    for (int nn = 0; nn < 4; ++nn) {
#pragma unroll
        for (int r = 0; r < 8; ++r) {
            int m = mrow + r;
            if (m < M) {
                int n = n0 + nn * 16 + ncol;
                float v = acc[nn][r];
                if (bias) v += bias[n];
                if (relu) v = fmaxf(v, 0.f);
                C[(size_t)m * Ncols + n] = (CT)v;
            }
        }
    }
}

// ------------------- weight f32 -> f16 conversion --------------------
__global__ void k_f32_to_f16(const float* __restrict__ s, h16* __restrict__ d, int n)
{
    int i = blockIdx.x * blockDim.x + threadIdx.x;
    if (i < n) d[i] = (h16)s[i];
}
// em_W1 (196x64) -> zero-padded (224x64)
__global__ void k_cvt_em1(const float* __restrict__ s, h16* __restrict__ d)
{
    int i = blockIdx.x * blockDim.x + threadIdx.x;
    if (i >= EIN_PAD * HID) return;
    d[i] = (i < EIN * HID) ? (h16)s[i] : (h16)0.f;
}

// ------------------- tree propagate (bottom-up) ----------------------
// S[node][0..12] = subtree leaf-feature sum, S[node][13] = leaf count.
__global__ void k_init_S(const float* __restrict__ nf, float* __restrict__ S)
{
    int idx = blockIdx.x * blockDim.x + threadIdx.x;
    if (idx >= NN * 14) return;
    int node = idx / 14, f = idx % 14;
    bool leaf = (node >= NIN);
    float v = 0.f;
    if (leaf) v = (f < NODE_F) ? nf[(size_t)node * NODE_F + f] : 1.f;
    S[idx] = v;
}
__global__ void k_level(float* __restrict__ S, int start, int count)
{
    int idx = blockIdx.x * blockDim.x + threadIdx.x;
    if (idx >= count * 14) return;
    int node = start + idx / 14, f = idx % 14;
    S[(size_t)node * 14 + f] += S[(size_t)(2 * node + 1) * 14 + f] +
                                S[(size_t)(2 * node + 2) * 14 + f];
}
__global__ void k_fill_x13(const float* __restrict__ S, const float* __restrict__ nf,
                           float* __restrict__ x13)
{
    int idx = blockIdx.x * blockDim.x + threadIdx.x;
    if (idx >= NN * NODE_F) return;
    int i = idx / NODE_F, f = idx % NODE_F;
    float v = nf[idx];
    if (i < NIN) {
        float cnt = S[(size_t)i * 14 + 13];
        if (cnt > 0.f) v = S[(size_t)i * 14 + f] / fmaxf(cnt, 1.f);
    }
    x13[idx] = v;
}

// ------------------- node MLP layer 1 (K=13, VALU) -------------------
__global__ void k_mlp1(const float* __restrict__ x13, const float* __restrict__ W1,
                       const float* __restrict__ b1, float* __restrict__ out)
{
    int idx = blockIdx.x * blockDim.x + threadIdx.x;
    if (idx >= NN * HID) return;
    int i = idx >> 6, j = idx & 63;
    float acc = b1[j];
    const float* xr = x13 + (size_t)i * NODE_F;
#pragma unroll
    for (int k = 0; k < NODE_F; ++k) acc += xr[k] * W1[k * HID + j];
    out[idx] = fmaxf(acc, 0.f);
}

// ------------------- GAT pieces --------------------------------------
// Incoming-edge source set of node i (tree structure + self loop):
__device__ __forceinline__ int node_neighbors(int i, int* s)
{
    int c = 0;
    s[c++] = i;                                 // self loop
    if (i > 0)   s[c++] = (i - 1) >> 1;         // parent
    if (i < NIN) { s[c++] = 2 * i + 1; s[c++] = 2 * i + 2; }  // children
    return c;
}

// xp is stored in f16 (halves the dominant activation traffic).
__global__ void k_gat_scores(const h16* __restrict__ xp, const float* __restrict__ asrc,
                             const float* __restrict__ adst, float* __restrict__ esed)
{
    int idx = blockIdx.x * blockDim.x + threadIdx.x;
    if (idx >= NN * HEADS) return;
    int i = idx >> 2, h = idx & 3;
    const h16* xr = xp + (size_t)i * (HEADS * HID) + h * HID;
    const float* as = asrc + h * HID;
    const float* ad = adst + h * HID;
    float es = 0.f, ed = 0.f;
#pragma unroll
    for (int f = 0; f < HID; ++f) {
        float xv = (float)xr[f];
        es += xv * as[f]; ed += xv * ad[f];
    }
    esed[(size_t)i * 8 + h]     = es;
    esed[(size_t)i * 8 + 4 + h] = ed;
}

__global__ void k_gat_alpha(const float* __restrict__ esed, float* __restrict__ alpha)
{
    int i = blockIdx.x * blockDim.x + threadIdx.x;
    if (i >= NN) return;
    int s[4]; int cnt = node_neighbors(i, s);
#pragma unroll
    for (int h = 0; h < HEADS; ++h) {
        float edv = esed[(size_t)i * 8 + 4 + h];
        float e[4]; float m = -1e30f;
        for (int j = 0; j < cnt; ++j) {
            float t = esed[(size_t)s[j] * 8 + h] + edv;
            t = (t > 0.f) ? t : NEG_SLOPE * t;   // leaky relu
            e[j] = t; m = fmaxf(m, t);
        }
        float den = 0.f;
        for (int j = 0; j < cnt; ++j) { e[j] = expf(e[j] - m); den += e[j]; }
        float inv = 1.f / den;
        for (int j = 0; j < cnt; ++j) alpha[(size_t)i * 16 + j * 4 + h] = e[j] * inv;
        for (int j = cnt; j < 4; ++j) alpha[(size_t)i * 16 + j * 4 + h] = 0.f;
    }
}

__global__ void k_gat_aggregate(const h16* __restrict__ xp, const float* __restrict__ alpha,
                                const float* __restrict__ xin, const float* __restrict__ gbias,
                                float* __restrict__ out)
{
    int idx = blockIdx.x * blockDim.x + threadIdx.x;
    if (idx >= NN * HID) return;
    int i = idx >> 6, f = idx & 63;
    int s[4]; int cnt = node_neighbors(i, s);
    float acc = 0.f;
    for (int j = 0; j < cnt; ++j) {
        const h16* xr = xp + (size_t)s[j] * (HEADS * HID);
        const float* al = alpha + (size_t)i * 16 + j * 4;
#pragma unroll
        for (int h = 0; h < HEADS; ++h) acc += al[h] * (float)xr[h * HID + f];
    }
    out[idx] = xin[idx] + acc * (1.f / HEADS) + gbias[f];
}

// LayerNorm over 64 features, one wave per node, wave32 shuffle reduction.
__global__ void k_layernorm(const float* __restrict__ t, const float* __restrict__ g,
                            const float* __restrict__ b, float* __restrict__ x)
{
    int node = (blockIdx.x * blockDim.x + threadIdx.x) >> 5;
    int lane = threadIdx.x & 31;
    if (node >= NN) return;
    float t0 = t[(size_t)node * HID + lane];
    float t1 = t[(size_t)node * HID + 32 + lane];
    float s = t0 + t1;
#pragma unroll
    for (int o = 16; o > 0; o >>= 1) s += __shfl_xor(s, o, 32);
    float mu = s * (1.f / HID);
    float d0 = t0 - mu, d1 = t1 - mu;
    float v = d0 * d0 + d1 * d1;
#pragma unroll
    for (int o = 16; o > 0; o >>= 1) v += __shfl_xor(v, o, 32);
    float rs = rsqrtf(v * (1.f / HID) + LN_EPS);
    x[(size_t)node * HID + lane]      = d0 * rs * g[lane]      + b[lane];
    x[(size_t)node * HID + 32 + lane] = d1 * rs * g[32 + lane] + b[32 + lane];
}

// --------------- edge-input assembly: [x[p], x[c], ef, gtf, 0pad] ----
// Assembled directly in f16: the edge GEMM consumes an f16 A-matrix.
__global__ void k_assemble_ein(const float* __restrict__ x, const int* __restrict__ ei,
                               const float* __restrict__ ef, const float* __restrict__ gtf,
                               h16* __restrict__ ein)
{
    int idx = blockIdx.x * blockDim.x + threadIdx.x;
    if (idx >= NE * EIN_PAD) return;
    int e = idx / EIN_PAD, c = idx % EIN_PAD;
    int p  = ei[2 * e];                 // edge_index[0, 2e]
    int ch = ei[2 * NE + 2 * e];        // edge_index[1, 2e]
    float v;
    if      (c < 64)  v = x[(size_t)p  * HID + c];
    else if (c < 128) v = x[(size_t)ch * HID + (c - 64)];
    else if (c < 132) v = ef[(size_t)e * 4 + (c - 128)];
    else if (c < 196) v = gtf[(size_t)e * HID + (c - 132)];
    else              v = 0.f;
    ein[idx] = (h16)v;
}

// --------------- head layer 2 (64 -> 2, VALU) ------------------------
__global__ void k_head2(const float* __restrict__ h1, const float* __restrict__ W2,
                        const float* __restrict__ b2, float* __restrict__ out)
{
    int idx = blockIdx.x * blockDim.x + threadIdx.x;
    if (idx >= NE * 2) return;
    int e = idx >> 1, o = idx & 1;
    float acc = b2[o];
    const float* hr = h1 + (size_t)e * HID;
#pragma unroll
    for (int f = 0; f < HID; ++f) acc += hr[f] * W2[f * 2 + o];
    out[idx] = acc;
}

// =====================================================================
static inline unsigned grid1(long long n) { return (unsigned)((n + 255) / 256); }
static inline unsigned gemm_grid(int M, int Ncols)
{
    long long waves = (long long)((M + 15) >> 4) * (Ncols >> 6);
    return (unsigned)((waves * 32 + 255) / 256);
}

extern "C" void kernel_launch(void* const* d_in, const int* in_sizes, int n_in,
                              void* d_out, int out_size, void* d_ws, size_t ws_size,
                              hipStream_t stream)
{
    (void)in_sizes; (void)n_in; (void)out_size; (void)ws_size;

    const float* nf    = (const float*)d_in[0];
    const int*   ei    = (const int*)  d_in[1];
    const float* ef    = (const float*)d_in[2];
    /* d_in[3] = is_leaf: unused, tree structure implies node >= NIN  */
    const float* gtf   = (const float*)d_in[4];
    const float* npW1  = (const float*)d_in[5];
    const float* npb1  = (const float*)d_in[6];
    const float* npW2  = (const float*)d_in[7];
    const float* npb2  = (const float*)d_in[8];
    const float* gatW  = (const float*)d_in[9];
    const float* gatAs = (const float*)d_in[10];
    const float* gatAd = (const float*)d_in[11];
    const float* gatB  = (const float*)d_in[12];
    const float* lng   = (const float*)d_in[13];
    const float* lnb   = (const float*)d_in[14];
    const float* emW1  = (const float*)d_in[15];
    const float* emb1  = (const float*)d_in[16];
    const float* emW2  = (const float*)d_in[17];
    const float* emb2  = (const float*)d_in[18];
    const float* hdW1  = (const float*)d_in[19];
    const float* hdb1  = (const float*)d_in[20];
    const float* hdW2  = (const float*)d_in[21];
    const float* hdb2  = (const float*)d_in[22];

    float* out_logits = (float*)d_out;                     // NE x 2
    float* out_emb    = (float*)d_out + (size_t)NE * 2;    // NE x 64

    // ---- workspace carve-up (256B aligned regions) ----
    char* wp = (char*)d_ws;
    auto walloc = [&](size_t bytes) -> void* {
        void* r = (void*)wp; wp += (bytes + 255) & ~(size_t)255; return r;
    };
    float* S     = (float*)walloc(sizeof(float) * (size_t)NN * 14);
    float* x13   = (float*)walloc(sizeof(float) * (size_t)NN * NODE_F);
    float* x     = (float*)walloc(sizeof(float) * (size_t)NN * HID);
    float* tmp   = (float*)walloc(sizeof(float) * (size_t)NN * HID);
    h16*   xp    = (h16*)  walloc(sizeof(h16)   * (size_t)NN * HEADS * HID); // also e_in (NE x 224 f16)
    float* esed  = (float*)walloc(sizeof(float) * (size_t)NN * 8);
    float* alpha = (float*)walloc(sizeof(float) * (size_t)NN * 16);
    h16* w_np2 = (h16*)walloc(sizeof(h16) * HID * HID);
    h16* w_gat = (h16*)walloc(sizeof(h16) * 3 * HID * HEADS * HID);
    h16* w_em1 = (h16*)walloc(sizeof(h16) * EIN_PAD * HID);
    h16* w_em2 = (h16*)walloc(sizeof(h16) * HID * HID);
    h16* w_hd1 = (h16*)walloc(sizeof(h16) * HID * HID);

    // ---- convert weights to f16 once per call ----
    k_f32_to_f16<<<grid1(HID * HID), 256, 0, stream>>>(npW2, w_np2, HID * HID);
    k_f32_to_f16<<<grid1(3 * HID * HEADS * HID), 256, 0, stream>>>(gatW, w_gat, 3 * HID * HEADS * HID);
    k_cvt_em1  <<<grid1(EIN_PAD * HID), 256, 0, stream>>>(emW1, w_em1);
    k_f32_to_f16<<<grid1(HID * HID), 256, 0, stream>>>(emW2, w_em2, HID * HID);
    k_f32_to_f16<<<grid1(HID * HID), 256, 0, stream>>>(hdW1, w_hd1, HID * HID);

    // ---- 1) bottom-up subtree-mean propagate ----
    k_init_S<<<grid1((long long)NN * 14), 256, 0, stream>>>(nf, S);
    for (int l = DEPTH - 2; l >= 0; --l) {
        int start = (1 << l) - 1;
        int count = 1 << l;
        k_level<<<grid1((long long)count * 14), 256, 0, stream>>>(S, start, count);
    }
    k_fill_x13<<<grid1((long long)NN * NODE_F), 256, 0, stream>>>(S, nf, x13);

    // ---- 2) node MLP ----
    k_mlp1<<<grid1((long long)NN * HID), 256, 0, stream>>>(x13, npW1, npb1, tmp);
    k_gemm<float, float><<<gemm_grid(NN, HID), 256, 0, stream>>>(
        tmp, NN, HID, w_np2, HID, npb2, x, 0);

    // ---- 3) three GAT layers ----
    for (int l = 0; l < 3; ++l) {
        k_gemm<float, h16><<<gemm_grid(NN, HEADS * HID), 256, 0, stream>>>(
            x, NN, HID, w_gat + (size_t)l * HID * HEADS * HID, HEADS * HID, nullptr, xp, 0);
        k_gat_scores<<<grid1((long long)NN * HEADS), 256, 0, stream>>>(
            xp, gatAs + l * HEADS * HID, gatAd + l * HEADS * HID, esed);
        k_gat_alpha<<<grid1(NN), 256, 0, stream>>>(esed, alpha);
        k_gat_aggregate<<<grid1((long long)NN * HID), 256, 0, stream>>>(
            xp, alpha, x, gatB + l * HID, tmp);
        k_layernorm<<<grid1((long long)NN * 32), 256, 0, stream>>>(
            tmp, lng + l * HID, lnb + l * HID, x);
    }

    // ---- 4) edge MLP (e_in assembled in f16, consumed as f16 A) ----
    h16* ein = xp;  // reuse xp region: NE*224 halves <= NN*256 halves
    k_assemble_ein<<<grid1((long long)NE * EIN_PAD), 256, 0, stream>>>(x, ei, ef, gtf, ein);
    k_gemm<h16, float><<<gemm_grid(NE, HID), 256, 0, stream>>>(
        ein, NE, EIN_PAD, w_em1, HID, emb1, tmp, 1);
    k_gemm<float, float><<<gemm_grid(NE, HID), 256, 0, stream>>>(
        tmp, NE, HID, w_em2, HID, emb2, out_emb, 0);

    // ---- 5) head ----
    k_gemm<float, float><<<gemm_grid(NE, HID), 256, 0, stream>>>(
        out_emb, NE, HID, w_hd1, HID, hdb1, tmp, 1);
    k_head2<<<grid1((long long)NE * 2), 256, 0, stream>>>(tmp, hdW2, hdb2, out_logits);
}
